// GIN_49194555408764
// MI455X (gfx1250) — compile-verified, matrix operand order checked by reading, not run on hardware
//
#include <hip/hip_runtime.h>
#include <hip/hip_bf16.h>

typedef __bf16 bf16;
typedef bf16  v16bf __attribute__((ext_vector_type(16)));
typedef bf16  v8bf  __attribute__((ext_vector_type(8)));
typedef float v8f   __attribute__((ext_vector_type(8)));

#define NODES  50000
#define EDGES  800000
#define FEAT   128
#define HID    256
#define GRAPHS 256
#define CLS    10
#define BNEPS  1e-5f

// ---- weight convert: w[K][N] f32 (row-major) -> wt[N][K] bf16 (row-major) ----
__global__ void k_wt(const float* __restrict__ w, bf16* __restrict__ wt, int K, int N) {
  int idx = blockIdx.x * blockDim.x + threadIdx.x;
  if (idx >= K * N) return;
  int n = idx / K, k = idx - n * K;
  wt[idx] = (bf16)w[(size_t)k * N + n];
}

// ---- edge gather + scatter-add: aggr[dst] += h[src], F = blockDim.x ----
__global__ void k_scatter(const float* __restrict__ h, const int* __restrict__ src,
                          const int* __restrict__ dst, float* __restrict__ aggr) {
  int e = blockIdx.x, f = threadIdx.x, F = blockDim.x;
  int s = src[e], d = dst[e];
  atomicAdd(&aggr[(size_t)d * F + f], h[(size_t)s * F + f]);
}

// ---- fused (1+eps)*x + aggr, convert to bf16 GEMM input ----
__global__ void k_preact(const float* __restrict__ h, const float* __restrict__ aggr,
                         const float* __restrict__ eps, int layer,
                         bf16* __restrict__ abf, int n) {
  int i = blockIdx.x * blockDim.x + threadIdx.x;
  if (i >= n) return;
  abf[i] = (bf16)((1.0f + eps[layer]) * h[i] + aggr[i]);
}

// ---- WMMA GEMM: A[M,K] bf16 row-major x Wt[N,K] bf16 row-major (B pre-transposed).
// N must be 256. Block = 128 threads = 4 waves; each wave computes a 16x64 strip
// (1 m-tile x 4 n-tiles), reusing the A fragment across 4 WMMAs per K-step.
// Epilogue: +bias -> ReLU -> optional BatchNorm; writes f32 and/or bf16 outputs.
__global__ __launch_bounds__(128) void k_gemm(
    const bf16* __restrict__ A, const bf16* __restrict__ Wt,
    const float* __restrict__ bias,
    const float* __restrict__ bn_g, const float* __restrict__ bn_b,
    const float* __restrict__ bn_m, const float* __restrict__ bn_v,
    float* __restrict__ outf, bf16* __restrict__ outb,
    int M, int K, int N)
{
  int wave = threadIdx.x >> 5, lane = threadIdx.x & 31;
  int mt = blockIdx.x;
  int nt0 = wave * 4;                    // 4 consecutive n-tiles per wave
  int half = lane >> 4, rc = lane & 15;

  // 16-bit A/B fragment layout (ISA 7.12.2): lane {half,rc} holds
  // K = half*8 + 0..7 (VGPR0-3) and K = 16 + half*8 + 0..7 (VGPR4-7) of row rc.
  const bf16* Ap = A  + (size_t)(mt * 16 + rc) * K + half * 8;
  const bf16* Bp = Wt + (size_t)(nt0 * 16 + rc) * K + half * 8;
  size_t bstep = (size_t)16 * K;         // next n-tile of B

  v8f acc[4] = {{}, {}, {}, {}};
  for (int k0 = 0; k0 < K; k0 += 32) {
    v16bf a;
    ((v8bf*)&a)[0] = *(const v8bf*)(Ap + k0);
    ((v8bf*)&a)[1] = *(const v8bf*)(Ap + k0 + 16);
    #pragma unroll
    for (int j = 0; j < 4; ++j) {
      v16bf b;
      ((v8bf*)&b)[0] = *(const v8bf*)(Bp + j * bstep + k0);
      ((v8bf*)&b)[1] = *(const v8bf*)(Bp + j * bstep + k0 + 16);
      acc[j] = __builtin_amdgcn_wmma_f32_16x16x32_bf16(false, a, false, b,
                                                       (short)0, acc[j], false, false);
    }
  }

  // C/D layout: lanes 0-15: VGPR r -> (M = r,     N = lane)
  //             lanes16-31: VGPR r -> (M = r + 8, N = lane - 16)
  int mbase = mt * 16 + half * 8;
  #pragma unroll
  for (int j = 0; j < 4; ++j) {
    int n = (nt0 + j) * 16 + rc;
    float bia = bias ? bias[n] : 0.0f;
    float scale = 1.0f, shift = 0.0f;
    bool bn = (bn_g != nullptr);
    if (bn) {
      float sc = bn_g[n] * rsqrtf(bn_v[n] + BNEPS);
      scale = sc;
      shift = bn_b[n] - bn_m[n] * sc;
    }
    #pragma unroll
    for (int r = 0; r < 8; ++r) {
      float v = acc[j][r] + bia;
      v = v > 0.0f ? v : 0.0f;           // ReLU
      if (bn) v = v * scale + shift;      // eval-mode BatchNorm
      size_t o = (size_t)(mbase + r) * N + n;
      if (outf) outf[o] = v;
      if (outb) outb[o] = (bf16)v;
    }
  }
}

// ---- pooling ----
__global__ void k_count(const int* __restrict__ batch, float* __restrict__ cnt) {
  int i = blockIdx.x * blockDim.x + threadIdx.x;
  if (i < NODES) atomicAdd(&cnt[batch[i]], 1.0f);
}
__global__ void k_poolsum(const float* __restrict__ h, const int* __restrict__ batch,
                          float* __restrict__ pooled) {
  int i = blockIdx.x, f = threadIdx.x;
  atomicAdd(&pooled[(size_t)batch[i] * HID + f], h[(size_t)i * HID + f]);
}
__global__ void k_poolmean(const float* __restrict__ pooled, const float* __restrict__ cnt,
                           bf16* __restrict__ pbf) {
  int idx = blockIdx.x * blockDim.x + threadIdx.x;   // GRAPHS*HID
  int g = idx >> 8;
  pbf[idx] = (bf16)(pooled[idx] / fmaxf(cnt[g], 1.0f));
}

// ---- lin2 (256x10) + log_softmax; one wave per graph ----
__global__ void k_head(const float* __restrict__ fc1, const float* __restrict__ w2,
                       const float* __restrict__ b2, float* __restrict__ out) {
  int g = blockIdx.x, c = threadIdx.x;
  __shared__ float lg[CLS];
  if (c < CLS) {
    float acc = b2[c];
    const float* row = fc1 + (size_t)g * HID;
    for (int k = 0; k < HID; ++k) acc += row[k] * w2[(size_t)k * CLS + c];
    lg[c] = acc;
  }
  __syncthreads();
  if (c < CLS) {
    float mx = lg[0];
    for (int i = 1; i < CLS; ++i) mx = fmaxf(mx, lg[i]);
    float s = 0.0f;
    for (int i = 0; i < CLS; ++i) s += __expf(lg[i] - mx);
    out[(size_t)g * CLS + c] = lg[c] - mx - __logf(s);
  }
}

extern "C" void kernel_launch(void* const* d_in, const int* in_sizes, int n_in,
                              void* d_out, int out_size, void* d_ws, size_t ws_size,
                              hipStream_t stream) {
  const float* x        = (const float*)d_in[0];
  const int*   ei       = (const int*)d_in[1];
  const int*   batch    = (const int*)d_in[2];
  const float* c1_w0    = (const float*)d_in[3];
  const float* c1_b0    = (const float*)d_in[4];
  const float* c1_w1    = (const float*)d_in[5];
  const float* c1_b1    = (const float*)d_in[6];
  const float* c1_g     = (const float*)d_in[7];
  const float* c1_be    = (const float*)d_in[8];
  const float* c1_m     = (const float*)d_in[9];
  const float* c1_v     = (const float*)d_in[10];
  const float* cs_w0    = (const float*)d_in[11];
  const float* cs_b0    = (const float*)d_in[12];
  const float* cs_w1    = (const float*)d_in[13];
  const float* cs_b1    = (const float*)d_in[14];
  const float* cs_g     = (const float*)d_in[15];
  const float* cs_be    = (const float*)d_in[16];
  const float* cs_m     = (const float*)d_in[17];
  const float* cs_v     = (const float*)d_in[18];
  const float* eps      = (const float*)d_in[19];
  const float* l1_w     = (const float*)d_in[20];
  const float* l1_b     = (const float*)d_in[21];
  const float* l2_w     = (const float*)d_in[22];
  const float* l2_b     = (const float*)d_in[23];
  const int* src = ei;
  const int* dst = ei + EDGES;
  float* out = (float*)d_out;

  // ---- workspace carve ----
  char* p = (char*)d_ws;
  auto alloc = [&](size_t bytes) { char* r = p; p += (bytes + 255) & ~(size_t)255; return r; };
  float* h      = (float*)alloc((size_t)NODES * HID * 4);
  float* aggr   = (float*)alloc((size_t)NODES * HID * 4);
  bf16*  abf    = (bf16*) alloc((size_t)NODES * HID * 2);
  bf16*  midbf  = (bf16*) alloc((size_t)NODES * HID * 2);
  bf16*  wt_c1w0 = (bf16*)alloc((size_t)FEAT * HID * 2);
  bf16*  wt_c1w1 = (bf16*)alloc((size_t)HID * HID * 2);
  bf16*  wt_cw0  = (bf16*)alloc((size_t)3 * HID * HID * 2);
  bf16*  wt_cw1  = (bf16*)alloc((size_t)3 * HID * HID * 2);
  bf16*  wt_l1   = (bf16*)alloc((size_t)HID * HID * 2);
  float* pooled = (float*)alloc((size_t)GRAPHS * HID * 4);
  float* cnt    = (float*)alloc((size_t)GRAPHS * 4);
  bf16*  pbf    = (bf16*) alloc((size_t)GRAPHS * HID * 2);
  float* fc1    = (float*)alloc((size_t)GRAPHS * HID * 4);

  // ---- weights -> bf16 transposed (N x K) ----
  k_wt<<<(FEAT * HID + 255) / 256, 256, 0, stream>>>(c1_w0, wt_c1w0, FEAT, HID);
  k_wt<<<(HID * HID + 255) / 256, 256, 0, stream>>>(c1_w1, wt_c1w1, HID, HID);
  for (int i = 0; i < 3; ++i) {
    k_wt<<<(HID * HID + 255) / 256, 256, 0, stream>>>(cs_w0 + (size_t)i * HID * HID,
                                                      wt_cw0 + (size_t)i * HID * HID, HID, HID);
    k_wt<<<(HID * HID + 255) / 256, 256, 0, stream>>>(cs_w1 + (size_t)i * HID * HID,
                                                      wt_cw1 + (size_t)i * HID * HID, HID, HID);
  }
  k_wt<<<(HID * HID + 255) / 256, 256, 0, stream>>>(l1_w, wt_l1, HID, HID);

  dim3 gblk(128);
  dim3 ggrid(NODES / 16);   // 3125 blocks; 4 waves x 4 n-tiles cover all 256 cols

  // ---- layer 1 (FEAT -> HID) ----
  hipMemsetAsync(aggr, 0, (size_t)NODES * FEAT * 4, stream);
  k_scatter<<<EDGES, FEAT, 0, stream>>>(x, src, dst, aggr);
  k_preact<<<(NODES * FEAT + 255) / 256, 256, 0, stream>>>(x, aggr, eps, 0, abf, NODES * FEAT);
  k_gemm<<<ggrid, gblk, 0, stream>>>(abf, wt_c1w0, c1_b0, nullptr, nullptr, nullptr, nullptr,
                                     nullptr, midbf, NODES, FEAT, HID);
  k_gemm<<<ggrid, gblk, 0, stream>>>(midbf, wt_c1w1, c1_b1, c1_g, c1_be, c1_m, c1_v,
                                     h, nullptr, NODES, HID, HID);

  // ---- layers 2..4 (HID -> HID) ----
  for (int i = 0; i < 3; ++i) {
    hipMemsetAsync(aggr, 0, (size_t)NODES * HID * 4, stream);
    k_scatter<<<EDGES, HID, 0, stream>>>(h, src, dst, aggr);
    k_preact<<<(NODES * HID + 255) / 256, 256, 0, stream>>>(h, aggr, eps, i + 1, abf, NODES * HID);
    k_gemm<<<ggrid, gblk, 0, stream>>>(abf, wt_cw0 + (size_t)i * HID * HID, cs_b0 + i * HID,
                                       nullptr, nullptr, nullptr, nullptr,
                                       nullptr, midbf, NODES, HID, HID);
    k_gemm<<<ggrid, gblk, 0, stream>>>(midbf, wt_cw1 + (size_t)i * HID * HID, cs_b1 + i * HID,
                                       cs_g + i * HID, cs_be + i * HID, cs_m + i * HID, cs_v + i * HID,
                                       h, nullptr, NODES, HID, HID);
  }

  // ---- global mean pool ----
  hipMemsetAsync(pooled, 0, (size_t)GRAPHS * HID * 4, stream);
  hipMemsetAsync(cnt, 0, (size_t)GRAPHS * 4, stream);
  k_count<<<(NODES + 255) / 256, 256, 0, stream>>>(batch, cnt);
  k_poolsum<<<NODES, HID, 0, stream>>>(h, batch, pooled);
  k_poolmean<<<(GRAPHS * HID) / 256, 256, 0, stream>>>(pooled, cnt, pbf);

  // ---- head: lin1 (WMMA, bias+ReLU) -> lin2 + log_softmax ----
  k_gemm<<<dim3(GRAPHS / 16), gblk, 0, stream>>>(pbf, wt_l1, l1_b,
                                                 nullptr, nullptr, nullptr, nullptr,
                                                 fc1, nullptr, GRAPHS, HID, HID);
  k_head<<<GRAPHS, 32, 0, stream>>>(fc1, l2_w, l2_b, out);
}
